// KroneckerAttention_38749194944563
// MI455X (gfx1250) — compile-verified, matrix operand order checked by reading, not run on hardware
//
#include <hip/hip_runtime.h>
#include <hip/hip_bf16.h>
#include <math.h>

// MI455X / gfx1250, wave32. Matrix math via V_WMMA_F32_16X16X32_BF16.
// NN GEMM: A tile + transposed/padded B tile staged in LDS so WMMA fragments
// are pure ds_load_b128; global side is coalesced global_load_b128/b64.
typedef __attribute__((ext_vector_type(16))) __bf16 v16bf;
typedef __attribute__((ext_vector_type(8)))  __bf16 v8bf;
typedef __attribute__((ext_vector_type(4)))  __bf16 v4bf;
typedef __attribute__((ext_vector_type(8)))  float  v8f;

// ---------------------------------------------------------------- utilities
__global__ void cvt_f32_bf16(const float* __restrict__ x, __bf16* __restrict__ y,
                             long long n) {
  long long i = (long long)blockIdx.x * blockDim.x + threadIdx.x;
  if (i < n) y[i] = (__bf16)x[i];
}

// LayerNorm over last dim d (256 or 512), one row per block (256 threads).
__global__ void ln_rows(const float* __restrict__ x, const float* __restrict__ g,
                        const float* __restrict__ b, __bf16* __restrict__ obf,
                        float* __restrict__ of, int d) {
  long long row = blockIdx.x;
  const float* xr = x + row * (long long)d;
  int tid = threadIdx.x;
  float s = 0.f, s2 = 0.f;
  for (int i = tid; i < d; i += blockDim.x) { float v = xr[i]; s += v; s2 += v * v; }
#pragma unroll
  for (int off = 16; off > 0; off >>= 1) {
    s  += __shfl_down(s,  off, 32);
    s2 += __shfl_down(s2, off, 32);
  }
  __shared__ float sh[18];
  int wv = tid >> 5;
  if ((tid & 31) == 0) { sh[wv] = s; sh[9 + wv] = s2; }
  __syncthreads();
  if (tid == 0) {
    float ts = 0.f, ts2 = 0.f;
    int nw = blockDim.x >> 5;
    for (int w = 0; w < nw; ++w) { ts += sh[w]; ts2 += sh[9 + w]; }
    float m = ts / d;
    sh[16] = m;
    sh[17] = ts2 / d - m * m;
  }
  __syncthreads();
  float m = sh[16];
  float inv = rsqrtf(sh[17] + 1e-5f);
  for (int i = tid; i < d; i += blockDim.x) {
    float v = (xr[i] - m) * inv * g[i] + b[i];
    long long o = row * (long long)d + i;
    if (obf) obf[o] = (__bf16)v;
    if (of)  of[o]  = v;
  }
}

// mean over axis 2 (W) or axis 1 (H) of u1:[B,H,W,d]
__global__ void mean_axis_k(const float* __restrict__ u1, float* __restrict__ out,
                            int B_, int H_, int W_, int d, int axis) {
  long long idx = (long long)blockIdx.x * blockDim.x + threadIdx.x;
  int Ho = (axis == 2) ? H_ : W_;
  long long tot = (long long)B_ * Ho * d;
  if (idx >= tot) return;
  int c  = (int)(idx % d);
  int r  = (int)((idx / d) % Ho);
  int bb = (int)(idx / ((long long)d * Ho));
  long long base, stride; int cnt;
  if (axis == 2) { // mean over W, r indexes H
    base = (((long long)bb * H_ + r) * W_) * d + c; stride = d; cnt = W_;
  } else {         // mean over H, r indexes W
    base = (long long)bb * H_ * W_ * d + (long long)r * d + c;
    stride = (long long)W_ * d; cnt = H_;
  }
  float sum = 0.f;
  for (int i = 0; i < cnt; ++i) sum += u1[base + (long long)i * stride];
  out[idx] = sum / (float)cnt;
}

// ---------------------------------------------------------------- WMMA GEMMs
// Fragment layouts per CDNA5 ISA §7.12.2 (wave32):
//   A 16x32 bf16 : lane = m + 16*khalf ; elems j<8 -> K=khalf*8+j, j>=8 -> 16+khalf*8+(j-8)
//   B 32x16 bf16 : lane = n + 16*khalf ; same K mapping (column-in-lane)
//   C/D 16x16 f32: lane = n + 16*g ; acc[rr] -> row rr + 8*g

#define BT_STRIDE 40   // padded LDS row stride for Bt (keeps 16B alignment)

// NN: C[z] = A[z/divA](MxK) * B[z](KxN, row-major) (+bias); f32/bf16 out.
// Block: 256 thr = 8 waves; tile M=32 x N=128; K step 32.
// LDS: As[32][32] natural, Bt[128][40] transposed -> all fragments ds_load_b128.
__global__ __launch_bounds__(256)
void wmma_gemm_nn(const __bf16* __restrict__ A, const __bf16* __restrict__ B,
                  const float* __restrict__ bias,
                  float* __restrict__ Cf, __bf16* __restrict__ Cb,
                  int M, int N, int K,
                  long long sA, long long sB, long long sC, int divA) {
  __shared__ __bf16 Bt[128 * BT_STRIDE];  // 10 KB
  __shared__ __bf16 As[32 * 32];          //  2 KB
  int z = blockIdx.z;
  const __bf16* Ab = A + (long long)(z / divA) * sA;
  const __bf16* Bb = B + (long long)z * sB;
  int tid  = threadIdx.x;
  int wave = tid >> 5;
  int lane = tid & 31;
  int half = lane >> 4;
  int r    = lane & 15;
  int nb = blockIdx.x * 128;
  int n0 = nb + wave * 16;
  int m0 = blockIdx.y * 32;

  // B cooperative mapping: each thread 8 cols x 2 k-rows (16 elems)
  int bcol = (tid & 15) * 8;     // 0..120
  int bk   = (tid >> 4) * 2;     // 0..30
  bool bload = (nb + bcol) < N;  // guard for N < 128 (T2 case)
  const __bf16* Bg = Bb + nb + bcol;

  // A cooperative mapping: each thread 1 row x 4 k (8 bytes)
  int arow = tid >> 3;           // 0..31
  int acol = (tid & 7) * 4;      // 0..28
  const __bf16* Ag = Ab + (long long)(m0 + arow) * K + acol;

  const __bf16* afrag0 = As + r * 32;           // rows m0 + r
  const __bf16* afrag1 = As + (16 + r) * 32;    // rows m0 + 16 + r
  const __bf16* bfragp = Bt + (wave * 16 + r) * BT_STRIDE;

  v8f acc0 = {}, acc1 = {};
  for (int k0 = 0; k0 < K; k0 += 32) {
    // stage A tile (one b64 per thread)
    *(v4bf*)(As + arow * 32 + acol) = *(const v4bf*)(Ag + k0);
    // stage B tile transposed (two b128 loads, 8 paired b32 stores)
    if (bload) {
      int boff = (k0 + bk) * N;
      v8bf r0 = *(const v8bf*)(Bg + boff);
      v8bf r1 = *(const v8bf*)(Bg + boff + N);
#pragma unroll
      for (int i = 0; i < 8; ++i) {
        __bf16* p = Bt + (bcol + i) * BT_STRIDE + bk;
        p[0] = r0[i];
        p[1] = r1[i];
      }
    }
    if (k0 + 32 < K) __builtin_prefetch(Ag + k0 + 32, 0, 3);
    __syncthreads();

    v8bf lo = *(const v8bf*)(afrag0 + half * 8);
    v8bf hi = *(const v8bf*)(afrag0 + 16 + half * 8);
    v16bf a0 = __builtin_shufflevector(lo, hi, 0,1,2,3,4,5,6,7,8,9,10,11,12,13,14,15);
    lo = *(const v8bf*)(afrag1 + half * 8);
    hi = *(const v8bf*)(afrag1 + 16 + half * 8);
    v16bf a1 = __builtin_shufflevector(lo, hi, 0,1,2,3,4,5,6,7,8,9,10,11,12,13,14,15);
    lo = *(const v8bf*)(bfragp + half * 8);
    hi = *(const v8bf*)(bfragp + 16 + half * 8);
    v16bf bfr = __builtin_shufflevector(lo, hi, 0,1,2,3,4,5,6,7,8,9,10,11,12,13,14,15);

    acc0 = __builtin_amdgcn_wmma_f32_16x16x32_bf16(
        false, a0, false, bfr, (short)0, acc0, false, false);
    acc1 = __builtin_amdgcn_wmma_f32_16x16x32_bf16(
        false, a1, false, bfr, (short)0, acc1, false, false);
    __syncthreads();
  }

  if (n0 < N) {
    int col = n0 + r;
    float bv = bias ? bias[col] : 0.f;
    long long cb = (long long)z * sC + col;
#pragma unroll
    for (int rr = 0; rr < 8; ++rr) {
      int row0 = m0 + rr + 8 * half;
      float v0 = acc0[rr] + bv;
      float v1 = acc1[rr] + bv;
      long long o0 = cb + (long long)row0 * N;
      long long o1 = o0 + (long long)16 * N;
      if (Cf) { Cf[o0] = v0; Cf[o1] = v1; }
      if (Cb) { Cb[o0] = (__bf16)v0; Cb[o1] = (__bf16)v1; }
    }
  }
}

// NT: C[z] = A[z](MxK) * B[z](NxK, row-major)^T -> bf16 out (used for q k^T).
// Both operands contiguous along K; no LDS needed.
__global__ __launch_bounds__(256)
void wmma_gemm_nt(const __bf16* __restrict__ A, const __bf16* __restrict__ B,
                  __bf16* __restrict__ Cb, int M, int N, int K,
                  long long sA, long long sB, long long sC) {
  int z = blockIdx.z;
  const __bf16* Ab = A + (long long)z * sA;
  const __bf16* Bb = B + (long long)z * sB;
  int wave = threadIdx.x >> 5;
  int lane = threadIdx.x & 31;
  int half = lane >> 4;
  int r    = lane & 15;
  int n0 = blockIdx.x * 128 + wave * 16;
  int m0 = blockIdx.y * 32;
  if (n0 >= N) return;   // wave-uniform, no barriers in this kernel

  const __bf16* Ar0 = Ab + (long long)(m0 + r) * K;
  const __bf16* Ar1 = Ab + (long long)(m0 + 16 + r) * K;
  const __bf16* Br  = Bb + (long long)(n0 + r) * K;

  v8f acc0 = {}, acc1 = {};
  for (int k0 = 0; k0 < K; k0 += 32) {
    v8bf lo = *(const v8bf*)(Ar0 + k0 + half * 8);
    v8bf hi = *(const v8bf*)(Ar0 + k0 + 16 + half * 8);
    v16bf a0 = __builtin_shufflevector(lo, hi, 0,1,2,3,4,5,6,7,8,9,10,11,12,13,14,15);
    lo = *(const v8bf*)(Ar1 + k0 + half * 8);
    hi = *(const v8bf*)(Ar1 + k0 + 16 + half * 8);
    v16bf a1 = __builtin_shufflevector(lo, hi, 0,1,2,3,4,5,6,7,8,9,10,11,12,13,14,15);
    lo = *(const v8bf*)(Br + k0 + half * 8);
    hi = *(const v8bf*)(Br + k0 + 16 + half * 8);
    v16bf bfr = __builtin_shufflevector(lo, hi, 0,1,2,3,4,5,6,7,8,9,10,11,12,13,14,15);
    acc0 = __builtin_amdgcn_wmma_f32_16x16x32_bf16(
        false, a0, false, bfr, (short)0, acc0, false, false);
    acc1 = __builtin_amdgcn_wmma_f32_16x16x32_bf16(
        false, a1, false, bfr, (short)0, acc1, false, false);
  }
  int col = n0 + r;
  long long cb = (long long)z * sC + col;
#pragma unroll
  for (int rr = 0; rr < 8; ++rr) {
    int row0 = m0 + rr + 8 * half;
    long long o0 = cb + (long long)row0 * N;
    Cb[o0] = (__bf16)acc0[rr];
    Cb[o0 + (long long)16 * N] = (__bf16)acc1[rr];
  }
}

// ------------------------------------------------------------------- RoPE
// q,k : [b, n, 8, 192] f32  ->  qo,ko : [b, 8, n, 192] bf16, rotated
__global__ void rope_qk(const float* __restrict__ q, const float* __restrict__ k,
                        __bf16* __restrict__ qo, __bf16* __restrict__ ko, int n) {
  int d = threadIdx.x;              // 0..191
  int idx = blockIdx.x;             // b*n*8
  int h   = idx % 8;
  int pos = (idx / 8) % n;
  int bb  = idx / (8 * n);
  float posv = (6.283185307179586f * 64.0f) * ((float)pos / (float)(n - 1));
  int fd = (d < 96) ? d : d - 96;
  float freq = posv * powf(10000.f, -((float)(2 * fd)) / 192.f);
  float c = cosf(freq), s = sinf(freq);
  long long src = (((long long)bb * n + pos) * 8 + h) * 192;
  long long dst = (((long long)bb * 8 + h) * n + pos) * 192 + d;
  float t   = q[src + d];
  float rot = (d < 96) ? -q[src + d + 96] : q[src + d - 96];
  qo[dst] = (__bf16)(t * c + rot * s);
  t   = k[src + d];
  rot = (d < 96) ? -k[src + d + 96] : k[src + d - 96];
  ko[dst] = (__bf16)(t * c + rot * s);
}

// v_bf [b,i,m,h,c] -> vp [b,h,i,(m*64+c)]
__global__ void permute_v(const __bf16* __restrict__ v, __bf16* __restrict__ vp) {
  long long idx = (long long)blockIdx.x * blockDim.x + threadIdx.x;
  const long long TOT = (long long)8 * 8 * 128 * 128 * 64;
  if (idx >= TOT) return;
  int c  = (int)(idx & 63);
  int m  = (int)((idx >> 6) & 127);
  int i  = (int)((idx >> 13) & 127);
  int h  = (int)((idx >> 20) & 7);
  int bb = (int)(idx >> 23);
  long long src = ((((long long)bb * 128 + i) * 128 + m) * 8 + h) * 64 + c;
  vp[idx] = v[src];
}

// out_p [b,h,i,l,c] f32 -> out_t [b,i,l,(h*64+c)] f32
__global__ void permute_out(const float* __restrict__ op, float* __restrict__ ot) {
  long long idx = (long long)blockIdx.x * blockDim.x + threadIdx.x;
  const long long TOT = (long long)8 * 128 * 128 * 512;
  if (idx >= TOT) return;
  int c  = (int)(idx & 63);
  int h  = (int)((idx >> 6) & 7);
  int l  = (int)((idx >> 9) & 127);
  int i  = (int)((idx >> 16) & 127);
  int bb = (int)(idx >> 23);
  long long src = ((((long long)bb * 8 + h) * 128 + i) * 128 + l) * 64 + c;
  ot[idx] = op[src];
}

// =========================================================================
extern "C" void kernel_launch(void* const* d_in, const int* in_sizes, int n_in,
                              void* d_out, int out_size, void* d_ws, size_t ws_size,
                              hipStream_t stream) {
  (void)in_sizes; (void)n_in; (void)out_size; (void)ws_size;
  const float* u        = (const float*)d_in[0];
  const float* norm_g   = (const float*)d_in[1];
  const float* norm_b   = (const float*)d_in[2];
  const float* Wf       = (const float*)d_in[3];
  const float* lnx_g    = (const float*)d_in[4];
  const float* lnx_b    = (const float*)d_in[5];
  const float* Wx1      = (const float*)d_in[6];
  const float* Wx2      = (const float*)d_in[7];
  const float* lny_g    = (const float*)d_in[8];
  const float* lny_b    = (const float*)d_in[9];
  const float* Wy1      = (const float*)d_in[10];
  const float* Wy2      = (const float*)d_in[11];
  const float* Wv       = (const float*)d_in[12];
  const float* Wq_x     = (const float*)d_in[13];
  const float* bq_x     = (const float*)d_in[14];
  const float* Wk_x     = (const float*)d_in[15];
  const float* bk_x     = (const float*)d_in[16];
  const float* Wq_y     = (const float*)d_in[17];
  const float* bq_y     = (const float*)d_in[18];
  const float* Wk_y     = (const float*)d_in[19];
  const float* bk_y     = (const float*)d_in[20];
  const float* out_ln_g = (const float*)d_in[21];
  const float* out_ln_b = (const float*)d_in[22];
  const float* Wo       = (const float*)d_in[23];

  const int Bz = 8, Hh = 128, Ww = 128, D = 256, HD = 512;
  const long long NT = (long long)Bz * Hh * Ww;           // 131072 tokens

  char* wp = (char*)d_ws;
  auto alloc = [&](size_t bytes) -> void* {
    void* r = (void*)wp;
    wp += (bytes + 255) & ~(size_t)255;
    return r;
  };
  auto cvt = [&](const float* src, __bf16* dst, long long n) {
    cvt_f32_bf16<<<(unsigned)((n + 255) / 256), 256, 0, stream>>>(src, dst, n);
  };
  auto gemm_nn = [&](const __bf16* A, const __bf16* Bm, const float* bias,
                     float* Cf, __bf16* Cb, int M, int N, int K,
                     int batch, long long sA, long long sB, long long sC, int divA) {
    dim3 grid((N + 127) / 128, M / 32, batch);
    wmma_gemm_nn<<<grid, 256, 0, stream>>>(A, Bm, bias, Cf, Cb, M, N, K,
                                           sA, sB, sC, divA);
  };
  auto gemm_nt = [&](const __bf16* A, const __bf16* Bm, __bf16* Cb,
                     int M, int N, int K, int batch,
                     long long sA, long long sB, long long sC) {
    dim3 grid((N + 127) / 128, M / 32, batch);
    wmma_gemm_nt<<<grid, 256, 0, stream>>>(A, Bm, Cb, M, N, K, sA, sB, sC);
  };

  // ---- weights -> bf16
  __bf16* Wf_b  = (__bf16*)alloc(65536 * 2);   cvt(Wf,  Wf_b,  65536);
  __bf16* Wv_b  = (__bf16*)alloc(131072 * 2);  cvt(Wv,  Wv_b,  131072);
  __bf16* Wx1_b = (__bf16*)alloc(65536 * 2);   cvt(Wx1, Wx1_b, 65536);
  __bf16* Wx2_b = (__bf16*)alloc(65536 * 2);   cvt(Wx2, Wx2_b, 65536);
  __bf16* Wy1_b = (__bf16*)alloc(65536 * 2);   cvt(Wy1, Wy1_b, 65536);
  __bf16* Wy2_b = (__bf16*)alloc(65536 * 2);   cvt(Wy2, Wy2_b, 65536);
  __bf16* Wqx_b = (__bf16*)alloc(393216 * 2);  cvt(Wq_x, Wqx_b, 393216);
  __bf16* Wkx_b = (__bf16*)alloc(393216 * 2);  cvt(Wk_x, Wkx_b, 393216);
  __bf16* Wqy_b = (__bf16*)alloc(393216 * 2);  cvt(Wq_y, Wqy_b, 393216);
  __bf16* Wky_b = (__bf16*)alloc(393216 * 2);  cvt(Wk_y, Wky_b, 393216);
  __bf16* Wo_b  = (__bf16*)alloc(131072 * 2);  cvt(Wo,  Wo_b,  131072);

  // ---- u1 = layernorm(u) @ Wf
  __bf16* u_ln = (__bf16*)alloc((size_t)NT * D * 2);
  ln_rows<<<(unsigned)NT, 256, 0, stream>>>(u, norm_g, norm_b, u_ln, nullptr, D);
  float*  u1  = (float*)alloc((size_t)NT * D * 4);
  __bf16* u1b = (__bf16*)alloc((size_t)NT * D * 2);
  gemm_nn(u_ln, Wf_b, nullptr, u1, u1b, (int)NT, D, D, 1, 0, 0, 0, 1);

  // ---- v = u1 @ Wv, then head-major permute
  __bf16* v_b = (__bf16*)alloc((size_t)NT * HD * 2);
  gemm_nn(u1b, Wv_b, nullptr, nullptr, v_b, (int)NT, HD, D, 1, 0, 0, 0, 1);
  __bf16* vp = (__bf16*)alloc((size_t)NT * HD * 2);
  {
    long long tot = NT * HD;
    permute_v<<<(unsigned)((tot + 255) / 256), 256, 0, stream>>>(v_b, vp);
  }

  // ---- axis means + layernorm + Wx1@Wx2 / Wy1@Wy2
  float* uxm = (float*)alloc(1024 * 256 * 4);
  float* uym = (float*)alloc(1024 * 256 * 4);
  {
    long long tot = (long long)Bz * Hh * D;
    mean_axis_k<<<(unsigned)((tot + 255) / 256), 256, 0, stream>>>(u1, uxm, Bz, Hh, Ww, D, 2);
    mean_axis_k<<<(unsigned)((tot + 255) / 256), 256, 0, stream>>>(u1, uym, Bz, Hh, Ww, D, 1);
  }
  __bf16* uxm_ln = (__bf16*)alloc(1024 * 256 * 2);
  __bf16* uym_ln = (__bf16*)alloc(1024 * 256 * 2);
  ln_rows<<<1024, 256, 0, stream>>>(uxm, lnx_g, lnx_b, uxm_ln, nullptr, D);
  ln_rows<<<1024, 256, 0, stream>>>(uym, lny_g, lny_b, uym_ln, nullptr, D);
  __bf16* tx1 = (__bf16*)alloc(1024 * 256 * 2);
  __bf16* ty1 = (__bf16*)alloc(1024 * 256 * 2);
  gemm_nn(uxm_ln, Wx1_b, nullptr, nullptr, tx1, 1024, D, D, 1, 0, 0, 0, 1);
  gemm_nn(uym_ln, Wy1_b, nullptr, nullptr, ty1, 1024, D, D, 1, 0, 0, 0, 1);
  __bf16* ux = (__bf16*)alloc(1024 * 256 * 2);
  __bf16* uy = (__bf16*)alloc(1024 * 256 * 2);
  gemm_nn(tx1, Wx2_b, nullptr, nullptr, ux, 1024, D, D, 1, 0, 0, 0, 1);
  gemm_nn(ty1, Wy2_b, nullptr, nullptr, uy, 1024, D, D, 1, 0, 0, 0, 1);

  // ---- q/k projections (+bias), RoPE, k1 = q k^T per (b,h)
  float* qx = (float*)alloc((size_t)1024 * 1536 * 4);
  float* kx = (float*)alloc((size_t)1024 * 1536 * 4);
  float* qy = (float*)alloc((size_t)1024 * 1536 * 4);
  float* ky = (float*)alloc((size_t)1024 * 1536 * 4);
  gemm_nn(ux, Wqx_b, bq_x, qx, nullptr, 1024, 1536, D, 1, 0, 0, 0, 1);
  gemm_nn(ux, Wkx_b, bk_x, kx, nullptr, 1024, 1536, D, 1, 0, 0, 0, 1);
  gemm_nn(uy, Wqy_b, bq_y, qy, nullptr, 1024, 1536, D, 1, 0, 0, 0, 1);
  gemm_nn(uy, Wky_b, bk_y, ky, nullptr, 1024, 1536, D, 1, 0, 0, 0, 1);

  __bf16* qxb = (__bf16*)alloc((size_t)64 * 128 * 192 * 2);
  __bf16* kxb = (__bf16*)alloc((size_t)64 * 128 * 192 * 2);
  __bf16* qyb = (__bf16*)alloc((size_t)64 * 128 * 192 * 2);
  __bf16* kyb = (__bf16*)alloc((size_t)64 * 128 * 192 * 2);
  rope_qk<<<8 * 128 * 8, 192, 0, stream>>>(qx, kx, qxb, kxb, 128);
  rope_qk<<<8 * 128 * 8, 192, 0, stream>>>(qy, ky, qyb, kyb, 128);

  __bf16* k1 = (__bf16*)alloc((size_t)64 * 128 * 128 * 2);
  __bf16* k2 = (__bf16*)alloc((size_t)64 * 128 * 128 * 2);
  gemm_nt(qxb, kxb, k1, 128, 128, 192, 64, 128 * 192, 128 * 192, 128 * 128);
  gemm_nt(qyb, kyb, k2, 128, 128, 192, 64, 128 * 192, 128 * 192, 128 * 128);

  // ---- Kronecker apply: T1[bh] = k1[bh] @ vp[bh]   (128 x 8192, K=128)
  __bf16* T1 = (__bf16*)alloc((size_t)64 * 128 * 8192 * 2);
  gemm_nn(k1, vp, nullptr, nullptr, T1, 128, 8192, 128, 64,
          (long long)128 * 128, (long long)128 * 8192, (long long)128 * 8192, 1);

  // ---- out_p[bh,i] = k2[bh] @ T1[bh,i]  (128 x 64, K=128), batch = bh*128+i
  float* outp = (float*)alloc((size_t)64 * 128 * 8192 * 4);
  gemm_nn(k2, T1, nullptr, outp, nullptr, 128, 64, 128, 64 * 128,
          (long long)128 * 128, (long long)8192, (long long)8192, 128);

  // ---- re-layout, final layernorm(512), @ Wo -> d_out (f32)
  float* outt = (float*)alloc((size_t)NT * HD * 4);
  {
    long long tot = NT * HD;
    permute_out<<<(unsigned)((tot + 255) / 256), 256, 0, stream>>>(outp, outt);
  }
  __bf16* outln = (__bf16*)alloc((size_t)NT * HD * 2);
  ln_rows<<<(unsigned)NT, 256, 0, stream>>>(outt, out_ln_g, out_ln_b, outln, nullptr, HD);
  gemm_nn(outln, Wo_b, nullptr, (float*)d_out, nullptr, (int)NT, 256, HD, 1, 0, 0, 0, 1);
}
